// AFormerLayerExpert_7155415515550
// MI455X (gfx1250) — matrix operand order repaired and checked by reading, not compiled, not run on hardware
//
#include <hip/hip_runtime.h>
#include <math.h>

// ---- problem constants -----------------------------------------------------
#define Bq   8
#define Sq   4096
#define Dq   768
#define Hq   12
#define Dhq  64
#define APAD 48          // 40 agents padded to 48 (3 x 16 tiles)
#define BSq  (Bq*Sq)     // 32768
#define BHq  (Bq*Hq)     // 96

typedef __bf16 bf16_t;
typedef __attribute__((ext_vector_type(16))) __bf16 v16bf;
typedef __attribute__((ext_vector_type(8)))  __bf16 v8bf;
typedef __attribute__((ext_vector_type(8)))  float  v8f;

__device__ inline bf16_t f2bf(float f) { return (bf16_t)f; }

// A fragment, 16x32 bf16, MxK. Per ISA table: lane L holds row m=L&15;
// half h=L>>4: VGPR0-3 = K[h*8 .. h*8+7], VGPR4-7 = K[16+h*8 .. 16+h*8+7].
__device__ inline v16bf frag_a(const bf16_t* A, int lda, int row0, int kbase, int lane) {
  int m = lane & 15, h = lane >> 4;
  const bf16_t* p = A + (size_t)(row0 + m) * lda + (kbase + h * 8);
  v8bf lo = *reinterpret_cast<const v8bf*>(p);
  v8bf hi = *reinterpret_cast<const v8bf*>(p + 16);
  return __builtin_shufflevector(lo, hi, 0,1,2,3,4,5,6,7,8,9,10,11,12,13,14,15);
}

// B fragment, 32x16 bf16 (KxN), loaded from an [N,K]-major buffer Bt:
// lane L holds column n=L&15; half h=L>>4 holds K[h*8 .. h*8+7], K[16+h*8 .. 16+h*8+7].
__device__ inline v16bf frag_b(const bf16_t* Bt, int ldb, int col0, int kbase, int lane) {
  int n = lane & 15, h = lane >> 4;
  const bf16_t* p = Bt + (size_t)(col0 + n) * ldb + (kbase + h * 8);
  v8bf lo = *reinterpret_cast<const v8bf*>(p);
  v8bf hi = *reinterpret_cast<const v8bf*>(p + 16);
  return __builtin_shufflevector(lo, hi, 0,1,2,3,4,5,6,7,8,9,10,11,12,13,14,15);
}

__device__ inline v8f wmma_bf16(v16bf a, v16bf b, v8f c) {
  return __builtin_amdgcn_wmma_f32_16x16x32_bf16(false, a, false, b, (short)0, c, false, false);
}

// 64x64 wave tile GEMM core: 16 WMMAs per k-step against 16 b128 loads
// -> 32 FLOP/byte from cache, double the 32x32 tile's intensity.
__device__ inline void gemm_tile64(const bf16_t* __restrict__ A, const bf16_t* __restrict__ Bt,
                                   int K, int lda, int ldb, int row0, int col0, int lane,
                                   v8f acc[4][4]) {
  for (int k = 0; k < K; k += 32) {
    v16bf a[4], b[4];
#pragma unroll
    for (int i = 0; i < 4; ++i) a[i] = frag_a(A, lda, row0 + 16 * i, k, lane);
#pragma unroll
    for (int j = 0; j < 4; ++j) b[j] = frag_b(Bt, ldb, col0 + 16 * j, k, lane);
    if (k + 128 < K) {   // prefetch A four k-steps ahead (global_prefetch_b8)
      const bf16_t* pf = A + (size_t)(row0 + (lane & 15)) * lda + (k + 128);
#pragma unroll
      for (int i = 0; i < 4; ++i) __builtin_prefetch(pf + (size_t)(16 * i) * lda, 0, 1);
    }
#pragma unroll
    for (int i = 0; i < 4; ++i)
#pragma unroll
      for (int j = 0; j < 4; ++j)
        acc[i][j] = wmma_bf16(a[i], b[j], acc[i][j]);
  }
}

// ---- elementwise converts --------------------------------------------------
__global__ void k_cvt_bf16(const float* __restrict__ in, bf16_t* __restrict__ out, int n) {
  int i = blockIdx.x * blockDim.x + threadIdx.x;
  if (i < n) out[i] = f2bf(in[i]);
}

// Wt[n*D + k] = W[k*D + n]  (weights to [N,K] bf16 for B-operand feed)
__global__ void k_cvt_wt(const float* __restrict__ W, bf16_t* __restrict__ Wt) {
  int i = blockIdx.x * blockDim.x + threadIdx.x;
  if (i < Dq * Dq) {
    int n = i / Dq, k = i % Dq;
    Wt[i] = f2bf(W[(size_t)k * Dq + n]);
  }
}

// ---- projection GEMM: C[BS,D] = A[BS,D] @ Wt^T + bias (bf16 out) ----------
__global__ __launch_bounds__(128) void k_gemm_proj(const bf16_t* __restrict__ A,
                                                   const bf16_t* __restrict__ Bt,
                                                   const float* __restrict__ bias,
                                                   bf16_t* __restrict__ C) {
  int lane = threadIdx.x & 31, wave = threadIdx.x >> 5;
  int wm = wave >> 1, wn = wave & 1;
  int row0 = blockIdx.y * 128 + wm * 64;
  int col0 = blockIdx.x * 128 + wn * 64;
  v8f acc[4][4] = {};
  gemm_tile64(A, Bt, Dq, Dq, Dq, row0, col0, lane, acc);
  int hl = lane >> 4, nn = lane & 15;
  for (int i = 0; i < 4; ++i)
    for (int j = 0; j < 4; ++j) {
      int c = col0 + j * 16 + nn;
      float bv = bias[c];
      int rbase = row0 + i * 16 + hl * 8;
      for (int r = 0; r < 8; ++r)
        C[(size_t)(rbase + r) * Dq + c] = f2bf(acc[i][j][r] + bv);
    }
}

// ---- V projection writing head-transposed vT[B,H,Dh,S] --------------------
__global__ __launch_bounds__(128) void k_gemm_projT(const bf16_t* __restrict__ A,
                                                    const bf16_t* __restrict__ Bt,
                                                    const float* __restrict__ bias,
                                                    bf16_t* __restrict__ vT) {
  int lane = threadIdx.x & 31, wave = threadIdx.x >> 5;
  int wm = wave >> 1, wn = wave & 1;
  int row0 = blockIdx.y * 128 + wm * 64;
  int col0 = blockIdx.x * 128 + wn * 64;
  v8f acc[4][4] = {};
  gemm_tile64(A, Bt, Dq, Dq, Dq, row0, col0, lane, acc);
  int hl = lane >> 4, nn = lane & 15;
  for (int i = 0; i < 4; ++i)
    for (int j = 0; j < 4; ++j) {
      int c = col0 + j * 16 + nn;      // feature index -> (head, dh)
      float bv = bias[c];
      int hh = c >> 6, dh = c & 63;
      int rbase = row0 + i * 16 + hl * 8;
      for (int r = 0; r < 8; ++r) {
        int m = rbase + r;             // token index -> (b, s)
        int b = m >> 12, s = m & (Sq - 1);
        vT[(((size_t)b * Hq + hh) * Dhq + dh) * Sq + s] = f2bf(acc[i][j][r] + bv);
      }
    }
}

// ---- agent tokens: bilinear interp of q rows along S; pad rows zeroed -----
__global__ void k_agents(const bf16_t* __restrict__ q, bf16_t* __restrict__ agents) {
  int ba = blockIdx.x; int b = ba / APAD, a = ba % APAD;
  int d = blockIdx.y * blockDim.x + threadIdx.x;
  bf16_t* o = agents + (size_t)ba * Dq + d;
  if (a >= 40) { *o = f2bf(0.f); return; }
  float src = (a + 0.5f) * ((float)Sq / 40.0f) - 0.5f;
  src = fminf(fmaxf(src, 0.f), (float)(Sq - 1));
  int lo = (int)floorf(src);
  int hi = min(lo + 1, Sq - 1);
  float w = src - (float)lo;
  float v0 = (float)q[((size_t)b * Sq + lo) * Dq + d];
  float v1 = (float)q[((size_t)b * Sq + hi) * Dq + d];
  *o = f2bf(v0 * (1.f - w) + w * v1);
}

// ---- stage 1 scores: sc[bh][m<48, s] = scale * agents · k^T ---------------
__global__ __launch_bounds__(128) void k_gemm_scores1(const bf16_t* __restrict__ agents,
                                                      const bf16_t* __restrict__ kbuf,
                                                      float* __restrict__ sc) {
  int lane = threadIdx.x & 31, wave = threadIdx.x >> 5;
  int bh = blockIdx.y, b = bh / Hq, h = bh % Hq;
  int s0 = blockIdx.x * 64 + wave * 16;
  const bf16_t* Ap = agents + (size_t)b * APAD * Dq + h * Dhq;
  const bf16_t* Bp = kbuf + (size_t)b * Sq * Dq + h * Dhq;   // row s, K contiguous
  v8f acc[3] = {};
  for (int k = 0; k < Dhq; k += 32) {
    v16bf bf = frag_b(Bp, Dq, s0, k, lane);
    v16bf a0 = frag_a(Ap, Dq, 0,  k, lane);
    v16bf a1 = frag_a(Ap, Dq, 16, k, lane);
    v16bf a2 = frag_a(Ap, Dq, 32, k, lane);
    acc[0] = wmma_bf16(a0, bf, acc[0]);
    acc[1] = wmma_bf16(a1, bf, acc[1]);
    acc[2] = wmma_bf16(a2, bf, acc[2]);
  }
  int hl = lane >> 4, nn = lane & 15;
  int s = s0 + nn;
  float* base = sc + (size_t)bh * APAD * Sq;
  for (int t = 0; t < 3; ++t)
    for (int r = 0; r < 8; ++r) {
      int m = t * 16 + hl * 8 + r;
      base[(size_t)m * Sq + s] = acc[t][r] * 0.125f;   // Dh^-0.5
    }
}

// ---- softmax over S per (bh, agent); pad rows -> zero probs ---------------
__global__ __launch_bounds__(256) void k_softmax1(const float* __restrict__ sc,
                                                  bf16_t* __restrict__ pr) {
  int bh = blockIdx.x / APAD, a = blockIdx.x % APAD;
  int tid = threadIdx.x;
  const float* row = sc + ((size_t)bh * APAD + a) * Sq;
  bf16_t* o = pr + ((size_t)bh * APAD + a) * Sq;
  if (a >= 40) { for (int i = tid; i < Sq; i += 256) o[i] = f2bf(0.f); return; }
  __shared__ float red[256];
  float m = -3.4e38f;
  for (int i = tid; i < Sq; i += 256) m = fmaxf(m, row[i]);
  red[tid] = m; __syncthreads();
  for (int s2 = 128; s2 > 0; s2 >>= 1) { if (tid < s2) red[tid] = fmaxf(red[tid], red[tid + s2]); __syncthreads(); }
  m = red[0]; __syncthreads();
  float sum = 0.f;
  for (int i = tid; i < Sq; i += 256) sum += __expf(row[i] - m);
  red[tid] = sum; __syncthreads();
  for (int s2 = 128; s2 > 0; s2 >>= 1) { if (tid < s2) red[tid] += red[tid + s2]; __syncthreads(); }
  float inv = 1.f / red[0];
  for (int i = tid; i < Sq; i += 256) o[i] = f2bf(__expf(row[i] - m) * inv);
}

// ---- agent_v = probs1 @ V, written transposed avT[bh][dh, agent64] --------
__global__ __launch_bounds__(128) void k_gemm_agentv(const bf16_t* __restrict__ pr,
                                                     const bf16_t* __restrict__ vT,
                                                     bf16_t* __restrict__ avT) {
  int lane = threadIdx.x & 31, wave = threadIdx.x >> 5;
  int bh = blockIdx.x;
  const bf16_t* Ap = pr + (size_t)bh * APAD * Sq;   // lda = S
  const bf16_t* Bp = vT + (size_t)bh * Dhq * Sq;    // row dh, K=s contiguous
  int n0 = wave * 16;
  v8f acc[3] = {};
  for (int k = 0; k < Sq; k += 32) {
    v16bf bf = frag_b(Bp, Sq, n0, k, lane);
    v16bf a0 = frag_a(Ap, Sq, 0,  k, lane);
    v16bf a1 = frag_a(Ap, Sq, 16, k, lane);
    v16bf a2 = frag_a(Ap, Sq, 32, k, lane);
    acc[0] = wmma_bf16(a0, bf, acc[0]);
    acc[1] = wmma_bf16(a1, bf, acc[1]);
    acc[2] = wmma_bf16(a2, bf, acc[2]);
  }
  int hl = lane >> 4, nn = lane & 15;
  int n = n0 + nn;
  bf16_t* base = avT + (size_t)bh * Dhq * 64;
  for (int t = 0; t < 3; ++t)
    for (int r = 0; r < 8; ++r) {
      int m = t * 16 + hl * 8 + r;
      base[(size_t)n * 64 + m] = f2bf(acc[t][r]);
    }
}

// ---- stage 2 scores: qsc[bh][s, a<48] = scale * q · agents^T --------------
__global__ __launch_bounds__(96) void k_gemm_scores2(const bf16_t* __restrict__ q,
                                                     const bf16_t* __restrict__ agents,
                                                     float* __restrict__ qsc) {
  int lane = threadIdx.x & 31, wave = threadIdx.x >> 5;   // 0..2 -> N subtile
  int bh = blockIdx.y, b = bh / Hq, h = bh % Hq;
  int s0 = blockIdx.x * 32;
  int n0 = wave * 16;
  const bf16_t* Ap = q + (size_t)b * Sq * Dq + h * Dhq;
  const bf16_t* Bp = agents + (size_t)b * APAD * Dq + h * Dhq;
  v8f acc[2] = {};
  for (int k = 0; k < Dhq; k += 32) {
    v16bf bf = frag_b(Bp, Dq, n0, k, lane);
    v16bf a0 = frag_a(Ap, Dq, s0,      k, lane);
    v16bf a1 = frag_a(Ap, Dq, s0 + 16, k, lane);
    acc[0] = wmma_bf16(a0, bf, acc[0]);
    acc[1] = wmma_bf16(a1, bf, acc[1]);
  }
  int hl = lane >> 4, nn = lane & 15;
  int a = n0 + nn;
  for (int t = 0; t < 2; ++t)
    for (int r = 0; r < 8; ++r) {
      int s = s0 + t * 16 + hl * 8 + r;
      qsc[((size_t)bh * Sq + s) * APAD + a] = acc[t][r] * 0.125f;
    }
}

// ---- softmax over 40 agents, write probs2[bh][s, 64] (cols>=40 zero) ------
__global__ __launch_bounds__(256) void k_softmax2(const float* __restrict__ qsc,
                                                  bf16_t* __restrict__ pr2) {
  int row = blockIdx.x * 8 + (threadIdx.x >> 5);
  int lane = threadIdx.x & 31;
  const float* r = qsc + (size_t)row * APAD;
  bf16_t* o = pr2 + (size_t)row * 64;
  float x0 = r[lane];                       // lanes 0..31 all < 40
  bool v1 = (lane + 32) < 40;
  float x1 = v1 ? r[lane + 32] : -3.4e38f;
  float m = fmaxf(x0, x1);
  for (int off = 16; off > 0; off >>= 1) m = fmaxf(m, __shfl_xor(m, off, 32));
  float e0 = __expf(x0 - m);
  float e1 = v1 ? __expf(x1 - m) : 0.f;
  float sum = e0 + e1;
  for (int off = 16; off > 0; off >>= 1) sum += __shfl_xor(sum, off, 32);
  float inv = 1.f / sum;
  o[lane] = f2bf(e0 * inv);
  o[lane + 32] = v1 ? f2bf(e1 * inv) : f2bf(0.f);
}

// ---- x = probs2 @ agent_v, written back to token layout xr[BS, D] ---------
__global__ __launch_bounds__(128) void k_gemm_x(const bf16_t* __restrict__ pr2,
                                                const bf16_t* __restrict__ avT,
                                                bf16_t* __restrict__ xr) {
  int lane = threadIdx.x & 31, wave = threadIdx.x >> 5;
  int wm = wave >> 1, wn = wave & 1;
  int bh = blockIdx.y, b = bh / Hq, h = bh % Hq;
  int s0 = blockIdx.x * 64 + wm * 32;
  int n0 = wn * 32;
  const bf16_t* Ap = pr2 + (size_t)bh * Sq * 64;   // lda = 64 (agent pad)
  const bf16_t* Bp = avT + (size_t)bh * Dhq * 64;  // row dh, K=agent contiguous
  v8f acc[2][2] = {};
  for (int k = 0; k < 64; k += 32) {
    v16bf a0 = frag_a(Ap, 64, s0,      k, lane);
    v16bf a1 = frag_a(Ap, 64, s0 + 16, k, lane);
    v16bf b0 = frag_b(Bp, 64, n0,      k, lane);
    v16bf b1 = frag_b(Bp, 64, n0 + 16, k, lane);
    acc[0][0] = wmma_bf16(a0, b0, acc[0][0]);
    acc[0][1] = wmma_bf16(a0, b1, acc[0][1]);
    acc[1][0] = wmma_bf16(a1, b0, acc[1][0]);
    acc[1][1] = wmma_bf16(a1, b1, acc[1][1]);
  }
  int hl = lane >> 4, nn = lane & 15;
  for (int i = 0; i < 2; ++i)
    for (int j = 0; j < 2; ++j) {
      int n = n0 + j * 16 + nn;
      int sbase = s0 + i * 16 + hl * 8;
      for (int r = 0; r < 8; ++r)
        xr[((size_t)b * Sq + sbase + r) * Dq + h * Dhq + n] = f2bf(acc[i][j][r]);
    }
}

// ---- output proj + residual (fp32 out) ------------------------------------
__global__ __launch_bounds__(128) void k_gemm_out(const bf16_t* __restrict__ A,
                                                  const bf16_t* __restrict__ Bt,
                                                  const float* __restrict__ bias,
                                                  const float* __restrict__ resid,
                                                  float* __restrict__ Y) {
  int lane = threadIdx.x & 31, wave = threadIdx.x >> 5;
  int wm = wave >> 1, wn = wave & 1;
  int row0 = blockIdx.y * 128 + wm * 64;
  int col0 = blockIdx.x * 128 + wn * 64;
  v8f acc[4][4] = {};
  gemm_tile64(A, Bt, Dq, Dq, Dq, row0, col0, lane, acc);
  int hl = lane >> 4, nn = lane & 15;
  for (int i = 0; i < 4; ++i)
    for (int j = 0; j < 4; ++j) {
      int c = col0 + j * 16 + nn;
      float bv = bias[c];
      int rbase = row0 + i * 16 + hl * 8;
      for (int r = 0; r < 8; ++r) {
        size_t idx = (size_t)(rbase + r) * Dq + c;
        Y[idx] = acc[i][j][r] + bv + resid[idx];
      }
    }
}

// ---- RMSNorm ---------------------------------------------------------------
__global__ __launch_bounds__(256) void k_rmsnorm(const float* __restrict__ Y,
                                                 const float* __restrict__ w,
                                                 float* __restrict__ out) {
  int row = blockIdx.x, tid = threadIdx.x;
  const float* p = Y + (size_t)row * Dq;
  __shared__ float red[256];
  float ss = 0.f;
  for (int i = tid; i < Dq; i += 256) { float v = p[i]; ss += v * v; }
  red[tid] = ss; __syncthreads();
  for (int s2 = 128; s2 > 0; s2 >>= 1) { if (tid < s2) red[tid] += red[tid + s2]; __syncthreads(); }
  float inv = rsqrtf(red[0] * (1.0f / Dq) + 1e-5f);
  for (int i = tid; i < Dq; i += 256) out[(size_t)row * Dq + i] = p[i] * inv * w[i];
}

// ---- host launcher ---------------------------------------------------------
extern "C" void kernel_launch(void* const* d_in, const int* in_sizes, int n_in,
                              void* d_out, int out_size, void* d_ws, size_t ws_size,
                              hipStream_t stream) {
  (void)in_sizes; (void)n_in; (void)out_size;
  const float* hs = (const float*)d_in[0];
  const float* Wq = (const float*)d_in[1];
  const float* bq = (const float*)d_in[2];
  const float* Wk = (const float*)d_in[3];
  const float* bk = (const float*)d_in[4];
  const float* Wv = (const float*)d_in[5];
  const float* bv = (const float*)d_in[6];
  const float* Wo = (const float*)d_in[7];
  const float* bo = (const float*)d_in[8];
  const float* nw = (const float*)d_in[9];
  float* out = (float*)d_out;

  // workspace layout (all sizes already 256B-aligned)
  const size_t sz_hsB = (size_t)BSq * Dq * 2;        // 50.3 MB (reused as xr)
  const size_t sz_WT  = (size_t)Dq * Dq * 2;         // 1.18 MB each
  const size_t sz_q   = sz_hsB;                      // q, then first half of y
  const size_t sz_k   = sz_hsB;                      // k, then second half of y
  const size_t sz_vT  = (size_t)BHq * Dhq * Sq * 2;  // 50.3 MB
  const size_t sz_ag  = (size_t)Bq * APAD * Dq * 2;  // 0.59 MB
  const size_t sz_avT = (size_t)BHq * Dhq * 64 * 2;  // 0.79 MB
  const size_t sz_sc  = (size_t)BHq * APAD * Sq * 4; // 75.5 MB (scores1, then qsc)
  const size_t sz_pr  = (size_t)BHq * Sq * 64 * 2;   // 50.3 MB (probs1, then probs2)

  char* w = (char*)d_ws;
  size_t off = 0;
  bf16_t* hsB = (bf16_t*)(w + off); off += sz_hsB;
  bf16_t* WqT = (bf16_t*)(w + off); off += sz_WT;
  bf16_t* WkT = (bf16_t*)(w + off); off += sz_WT;
  bf16_t* WvT = (bf16_t*)(w + off); off += sz_WT;
  bf16_t* WoT = (bf16_t*)(w + off); off += sz_WT;
  bf16_t* qb  = (bf16_t*)(w + off); off += sz_q;
  bf16_t* kb  = (bf16_t*)(w + off); off += sz_k;
  bf16_t* vT  = (bf16_t*)(w + off); off += sz_vT;
  bf16_t* ag  = (bf16_t*)(w + off); off += sz_ag;
  bf16_t* avT = (bf16_t*)(w + off); off += sz_avT;
  float*  sc  = (float*) (w + off); off += sz_sc;   // scores1 / qsc
  bf16_t* pr  = (bf16_t*)(w + off); off += sz_pr;   // probs1 / probs2
  if (ws_size < off) return;
  bf16_t* xr = hsB;                  // reuse: hsB dead after projections
  float*  y  = (float*)qb;           // reuse: q+k span exactly BS*D*4 bytes

  // 1. convert activations and weights to bf16 ([N,K] for weights)
  k_cvt_bf16<<<(BSq * Dq) / 256, 256, 0, stream>>>(hs, hsB, BSq * Dq);
  k_cvt_wt<<<(Dq * Dq + 255) / 256, 256, 0, stream>>>(Wq, WqT);
  k_cvt_wt<<<(Dq * Dq + 255) / 256, 256, 0, stream>>>(Wk, WkT);
  k_cvt_wt<<<(Dq * Dq + 255) / 256, 256, 0, stream>>>(Wv, WvT);
  k_cvt_wt<<<(Dq * Dq + 255) / 256, 256, 0, stream>>>(Wo, WoT);

  // 2. Q/K/V projections (WMMA, 64x64 wave tiles, 128x128 block tiles)
  dim3 gProj(Dq / 128, BSq / 128);
  k_gemm_proj <<<gProj, 128, 0, stream>>>(hsB, WqT, bq, qb);
  k_gemm_proj <<<gProj, 128, 0, stream>>>(hsB, WkT, bk, kb);
  k_gemm_projT<<<gProj, 128, 0, stream>>>(hsB, WvT, bv, vT);

  // 3. agent tokens (interp along S)
  k_agents<<<dim3(Bq * APAD, Dq / 256), 256, 0, stream>>>(qb, ag);

  // 4. stage 1: agents attend over K/V
  k_gemm_scores1<<<dim3(Sq / 64, BHq), 128, 0, stream>>>(ag, kb, sc);
  k_softmax1<<<BHq * APAD, 256, 0, stream>>>(sc, pr);
  k_gemm_agentv<<<BHq, 128, 0, stream>>>(pr, vT, avT);

  // 5. stage 2: queries attend over agents
  k_gemm_scores2<<<dim3(Sq / 32, BHq), 96, 0, stream>>>(qb, ag, sc);
  k_softmax2<<<(BHq * Sq) / 8, 256, 0, stream>>>(sc, pr);
  k_gemm_x<<<dim3(Sq / 64, BHq), 128, 0, stream>>>(pr, avT, xr);

  // 6. output projection + residual, then RMSNorm
  k_gemm_out<<<gProj, 128, 0, stream>>>(xr, WoT, bo, hs, y);
  k_rmsnorm<<<BSq, 256, 0, stream>>>(y, nw, out);
}
